// ExpertParallelMoE_5927054868630
// MI455X (gfx1250) — matrix operand order repaired and checked by reading, not compiled, not run on hardware
//
#include <hip/hip_runtime.h>
#include <hip/hip_bf16.h>
#include <math.h>

// Problem constants (match reference)
#define BT      8192      // B*S tokens
#define DMODEL  1024
#define DFF     2048
#define NEXP    8
#define NSLOT   (BT * 2)  // top-2 -> exactly 2 slots per token

typedef __attribute__((ext_vector_type(16))) __bf16        v16bf;
typedef __attribute__((ext_vector_type(8)))  float         v8f;
typedef __attribute__((ext_vector_type(8)))  unsigned int  v8u;

#define LDT 34   // LDS row stride in bf16 units: 68B = 17 banks -> conflict-free frag reads

static __device__ __forceinline__ unsigned short f2bf(float f) {
    unsigned int u = __builtin_bit_cast(unsigned int, f);
    unsigned int r = u + 0x7FFFu + ((u >> 16) & 1u);   // round-to-nearest-even
    return (unsigned short)(r >> 16);
}

static __device__ __forceinline__ float gelu_tanh(float x) {
    // jax.nn.gelu default (approximate=True)
    float x3 = x * x * x;
    return 0.5f * x * (1.0f + tanhf(0.7978845608028654f * (x + 0.044715f * x3)));
}

// CDNA5 async global->LDS copy (ASYNCcnt-tracked). lds_off = low 32 bits of the
// generic LDS pointer (== byte offset into LDS per flat-aperture rules).
static __device__ __forceinline__ void async_b64(unsigned int lds_off, const void* g) {
    asm volatile("global_load_async_to_lds_b64 %0, %1, off"
                 :: "v"(lds_off), "v"((unsigned long long)(uintptr_t)g)
                 : "memory");
}
static __device__ __forceinline__ void wait_async0() {
    asm volatile("s_wait_asynccnt 0x0" ::: "memory");
}

// ---------------------------------------------------------------- routing ---
__global__ void moe_zero(int* counts) {
    if (threadIdx.x < NEXP) counts[threadIdx.x] = 0;
}

// wave-per-token router: logits = h[t] @ router_w, top-2 gates = softmax of top-2 logits
__global__ void moe_router(const float* __restrict__ h, const float* __restrict__ rw,
                           int* __restrict__ counts, int* __restrict__ tok_idx,
                           float* __restrict__ tok_gate) {
    const int lane = threadIdx.x & 31;
    const int t = blockIdx.x * (blockDim.x >> 5) + (threadIdx.x >> 5);
    if (t >= BT) return;

    float acc[NEXP];
#pragma unroll
    for (int e = 0; e < NEXP; ++e) acc[e] = 0.0f;

    const float* hr = h + (size_t)t * DMODEL;
    for (int d = lane; d < DMODEL; d += 32) {
        const float hv = hr[d];
#pragma unroll
        for (int e = 0; e < NEXP; ++e) acc[e] += hv * rw[d * NEXP + e];
    }
#pragma unroll
    for (int e = 0; e < NEXP; ++e) {
#pragma unroll
        for (int off = 16; off > 0; off >>= 1)
            acc[e] += __shfl_xor(acc[e], off, 32);
    }
    if (lane == 0) {
        int i0 = 0;
#pragma unroll
        for (int e = 1; e < NEXP; ++e) if (acc[e] > acc[i0]) i0 = e;   // ties -> lowest idx
        int i1 = (i0 == 0) ? 1 : 0;
#pragma unroll
        for (int e = 0; e < NEXP; ++e)
            if (e != i0 && acc[e] > acc[i1]) i1 = e;
        const float g  = __expf(acc[i1] - acc[i0]);
        const float g0 = 1.0f / (1.0f + g);
        tok_idx[t * 2 + 0]  = i0;
        tok_idx[t * 2 + 1]  = i1;
        tok_gate[t * 2 + 0] = g0;
        tok_gate[t * 2 + 1] = 1.0f - g0;
        atomicAdd(&counts[i0], 1);
        atomicAdd(&counts[i1], 1);
    }
}

__global__ void moe_offsets(const int* __restrict__ counts, int* __restrict__ bases,
                            int* __restrict__ cursors) {
    if (threadIdx.x == 0) {
        int s = 0;
        for (int e = 0; e < NEXP; ++e) { bases[e] = s; cursors[e] = s; s += counts[e]; }
    }
}

__global__ void moe_scatter(const int* __restrict__ tok_idx, int* __restrict__ cursors,
                            int* __restrict__ slot_token, int* __restrict__ slot_of) {
    const int i = blockIdx.x * blockDim.x + threadIdx.x;
    if (i >= NSLOT) return;
    const int e = tok_idx[i];
    const int pos = atomicAdd(&cursors[e], 1);
    slot_token[pos] = i >> 1;
    slot_of[i] = pos;
}

// --------------------------------------------------- precision conversion ---
// elementwise fp32 -> bf16 (n multiple of 4)
__global__ void conv_bf16(const float* __restrict__ in, unsigned short* __restrict__ out,
                          int n) {
    const int i = (blockIdx.x * blockDim.x + threadIdx.x) * 4;
    if (i >= n) return;
    const float4 v = *(const float4*)(in + i);
    uint2 o;
    o.x = (unsigned int)f2bf(v.x) | ((unsigned int)f2bf(v.y) << 16);
    o.y = (unsigned int)f2bf(v.z) | ((unsigned int)f2bf(v.w) << 16);
    *(uint2*)(out + i) = o;
}

// per-expert transpose + convert: in [E][K][N] f32 -> out [E][N][K] bf16 (64x64 tiles)
__global__ __launch_bounds__(256)
void transp_bf16(const float* __restrict__ in, unsigned short* __restrict__ out,
                 int K, int N) {
    __shared__ unsigned short t[64][65];
    const int e  = blockIdx.z;
    const int k0 = blockIdx.y * 64;
    const int n0 = blockIdx.x * 64;
    const float* ine = in + (size_t)e * K * N;
    unsigned short* oute = out + (size_t)e * K * N;

    const int tx = threadIdx.x & 63;   // along N (coalesced read)
    const int ty = threadIdx.x >> 6;   // 0..3
#pragma unroll
    for (int it = 0; it < 16; ++it) {
        const int k = it * 4 + ty;
        t[tx][k] = f2bf(ine[(size_t)(k0 + k) * N + n0 + tx]);
    }
    __syncthreads();
    const int nx = threadIdx.x >> 2;          // output row (N index)
    const int kx = (threadIdx.x & 3) * 16;    // 16 bf16 = 32B contiguous
    unsigned int* dst = (unsigned int*)(oute + (size_t)(n0 + nx) * K + k0 + kx);
#pragma unroll
    for (int j = 0; j < 8; ++j)
        dst[j] = (unsigned int)t[nx][kx + 2 * j] |
                 ((unsigned int)t[nx][kx + 2 * j + 1] << 16);
}

// ---------------------------------------------------- WMMA fragment helpers ---
// A/B 16-bit fragments per ISA 7.12.2 (wave32).
static __device__ __forceinline__ v16bf load_frag_A(const unsigned short* __restrict__ lds,
                                                    int rowBase, int l16, int half) {
    const unsigned int* row = (const unsigned int*)(lds + (size_t)(rowBase + l16) * LDT);
    v8u u;
#pragma unroll
    for (int i = 0; i < 8; ++i) {
        const int k = ((i >> 2) << 4) + (half << 3) + ((i & 3) << 1); // K of dword i
        u[i] = row[k >> 1];
    }
    return __builtin_bit_cast(v16bf, u);
}

static __device__ __forceinline__ v16bf load_frag_B(const unsigned short* __restrict__ lds,
                                                    int colBase, int l16, int half) {
    const unsigned int* row = (const unsigned int*)(lds + (size_t)(colBase + l16) * LDT);
    v8u u;
#pragma unroll
    for (int i = 0; i < 8; ++i) u[i] = row[((i >> 2) << 3) + (half << 2) + (i & 3)]; // K = (i>>2)*16 + half*8 + 2*(i&3)
    return __builtin_bit_cast(v16bf, u);
}

// ============================ fast path (bf16 pre-converted, async staging) ==

// hid[slot] = gelu( hbf[token(slot)] @ w1t[e]^T + b1[e] )  (w1t is [e][DFF][DMODEL])
__global__ __launch_bounds__(256)
void moe_gemm1_bf(const unsigned short* __restrict__ hbf, const unsigned short* __restrict__ w1t,
                  const float* __restrict__ b1, const int* __restrict__ counts,
                  const int* __restrict__ bases, const int* __restrict__ slot_token,
                  unsigned short* __restrict__ hid) {
    const int e     = blockIdx.z;
    const int cnt   = counts[e];
    const int mBase = blockIdx.y * 128;
    if (mBase >= cnt) return;
    const int sBase = bases[e];
    const int nBase = blockIdx.x * 128;

    __shared__ unsigned short ldsA[128 * LDT];
    __shared__ unsigned short ldsB[128 * LDT];

    const int tid  = threadIdx.x;
    const int lane = tid & 31;
    const int wave = tid >> 5;
    const int wm   = (wave & 3) * 32;
    const int wn   = (wave >> 2) * 64;
    const int half = lane >> 4;
    const int l16  = lane & 15;

    v8f acc[2][4];
#pragma unroll
    for (int i = 0; i < 2; ++i)
#pragma unroll
        for (int j = 0; j < 4; ++j) acc[i][j] = (v8f){0.f,0.f,0.f,0.f,0.f,0.f,0.f,0.f};

    const int lastSlot = sBase + cnt - 1;
    const int aRow = tid >> 3;                 // 0..31
    const int aK4  = (tid & 7) * 4;            // 0..28 (4 bf16 = 8B)
    const unsigned int ldsAoff = (unsigned int)(uintptr_t)ldsA;
    const unsigned int ldsBoff = (unsigned int)(uintptr_t)ldsB;

    const unsigned short* aptr[4];
    const unsigned short* bptr[4];
    unsigned int aoff[4], boff[4];
#pragma unroll
    for (int it = 0; it < 4; ++it) {
        int slot = sBase + mBase + aRow + it * 32;
        slot = (slot > lastSlot) ? lastSlot : slot;
        aptr[it] = hbf + (size_t)slot_token[slot] * DMODEL + aK4;
        bptr[it] = w1t + ((size_t)e * DFF + nBase + aRow + it * 32) * DMODEL + aK4;
        aoff[it] = ldsAoff + (unsigned int)(((aRow + it * 32) * LDT + aK4) * 2);
        boff[it] = ldsBoff + (unsigned int)(((aRow + it * 32) * LDT + aK4) * 2);
    }

    for (int kc = 0; kc < DMODEL / 32; ++kc) {
        const int k0 = kc * 32;
#pragma unroll
        for (int it = 0; it < 4; ++it) async_b64(aoff[it], aptr[it] + k0);
#pragma unroll
        for (int it = 0; it < 4; ++it) async_b64(boff[it], bptr[it] + k0);
        wait_async0();
        __syncthreads();

        v16bf afr[2], bfr[4];
#pragma unroll
        for (int mi = 0; mi < 2; ++mi) afr[mi] = load_frag_A(ldsA, wm + mi * 16, l16, half);
#pragma unroll
        for (int ni = 0; ni < 4; ++ni) bfr[ni] = load_frag_B(ldsB, wn + ni * 16, l16, half);
#pragma unroll
        for (int mi = 0; mi < 2; ++mi)
#pragma unroll
            for (int ni = 0; ni < 4; ++ni)
                acc[mi][ni] = __builtin_amdgcn_wmma_f32_16x16x32_bf16(
                    false, afr[mi], false, bfr[ni], (short)0, acc[mi][ni], false, false);
        __syncthreads();
    }

#pragma unroll
    for (int mi = 0; mi < 2; ++mi) {
#pragma unroll
        for (int ni = 0; ni < 4; ++ni) {
            const int nG = nBase + wn + ni * 16 + l16;
            const float bias = b1[e * DFF + nG];
#pragma unroll
            for (int r = 0; r < 8; ++r) {
                const int slot = sBase + mBase + wm + mi * 16 + half * 8 + r;
                if (slot <= lastSlot)
                    hid[(size_t)slot * DFF + nG] = f2bf(gelu_tanh(acc[mi][ni][r] + bias));
            }
        }
    }
}

// y[slot] = hid[slot] @ w2t[e]^T + b2[e]   (w2t is [e][DMODEL][DFF])
__global__ __launch_bounds__(256)
void moe_gemm2_bf(const unsigned short* __restrict__ hid, const unsigned short* __restrict__ w2t,
                  const float* __restrict__ b2, const int* __restrict__ counts,
                  const int* __restrict__ bases, float* __restrict__ ybuf) {
    const int e     = blockIdx.z;
    const int cnt   = counts[e];
    const int mBase = blockIdx.y * 128;
    if (mBase >= cnt) return;
    const int sBase = bases[e];
    const int nBase = blockIdx.x * 128;

    __shared__ unsigned short ldsA[128 * LDT];
    __shared__ unsigned short ldsB[128 * LDT];

    const int tid  = threadIdx.x;
    const int lane = tid & 31;
    const int wave = tid >> 5;
    const int wm   = (wave & 3) * 32;
    const int wn   = (wave >> 2) * 64;
    const int half = lane >> 4;
    const int l16  = lane & 15;

    v8f acc[2][4];
#pragma unroll
    for (int i = 0; i < 2; ++i)
#pragma unroll
        for (int j = 0; j < 4; ++j) acc[i][j] = (v8f){0.f,0.f,0.f,0.f,0.f,0.f,0.f,0.f};

    const int lastSlot = sBase + cnt - 1;
    const int aRow = tid >> 3;
    const int aK4  = (tid & 7) * 4;
    const unsigned int ldsAoff = (unsigned int)(uintptr_t)ldsA;
    const unsigned int ldsBoff = (unsigned int)(uintptr_t)ldsB;

    const unsigned short* aptr[4];
    const unsigned short* bptr[4];
    unsigned int aoff[4], boff[4];
#pragma unroll
    for (int it = 0; it < 4; ++it) {
        int slot = sBase + mBase + aRow + it * 32;
        slot = (slot > lastSlot) ? lastSlot : slot;
        aptr[it] = hid + (size_t)slot * DFF + aK4;
        bptr[it] = w2t + ((size_t)e * DMODEL + nBase + aRow + it * 32) * DFF + aK4;
        aoff[it] = ldsAoff + (unsigned int)(((aRow + it * 32) * LDT + aK4) * 2);
        boff[it] = ldsBoff + (unsigned int)(((aRow + it * 32) * LDT + aK4) * 2);
    }

    for (int kc = 0; kc < DFF / 32; ++kc) {
        const int k0 = kc * 32;
#pragma unroll
        for (int it = 0; it < 4; ++it) async_b64(aoff[it], aptr[it] + k0);
#pragma unroll
        for (int it = 0; it < 4; ++it) async_b64(boff[it], bptr[it] + k0);
        wait_async0();
        __syncthreads();

        v16bf afr[2], bfr[4];
#pragma unroll
        for (int mi = 0; mi < 2; ++mi) afr[mi] = load_frag_A(ldsA, wm + mi * 16, l16, half);
#pragma unroll
        for (int ni = 0; ni < 4; ++ni) bfr[ni] = load_frag_B(ldsB, wn + ni * 16, l16, half);
#pragma unroll
        for (int mi = 0; mi < 2; ++mi)
#pragma unroll
            for (int ni = 0; ni < 4; ++ni)
                acc[mi][ni] = __builtin_amdgcn_wmma_f32_16x16x32_bf16(
                    false, afr[mi], false, bfr[ni], (short)0, acc[mi][ni], false, false);
        __syncthreads();
    }

#pragma unroll
    for (int mi = 0; mi < 2; ++mi) {
#pragma unroll
        for (int ni = 0; ni < 4; ++ni) {
            const int nG = nBase + wn + ni * 16 + l16;
            const float bias = b2[e * DMODEL + nG];
#pragma unroll
            for (int r = 0; r < 8; ++r) {
                const int slot = sBase + mBase + wm + mi * 16 + half * 8 + r;
                if (slot <= lastSlot)
                    ybuf[(size_t)slot * DMODEL + nG] = acc[mi][ni][r] + bias;
            }
        }
    }
}

// ===================== fallback path (fp32 inputs, in-loop conversion) ======

__global__ __launch_bounds__(256)
void moe_gemm1(const float* __restrict__ h, const float* __restrict__ w1,
               const float* __restrict__ b1, const int* __restrict__ counts,
               const int* __restrict__ bases, const int* __restrict__ slot_token,
               unsigned short* __restrict__ hid) {
    const int e     = blockIdx.z;
    const int cnt   = counts[e];
    const int mBase = blockIdx.y * 128;
    if (mBase >= cnt) return;
    const int sBase = bases[e];
    const int nBase = blockIdx.x * 128;
    const float* w1e = w1 + (size_t)e * DMODEL * DFF;

    __shared__ unsigned short ldsA[128 * LDT];
    __shared__ unsigned short ldsB[128 * LDT];

    const int tid  = threadIdx.x;
    const int lane = tid & 31;
    const int wave = tid >> 5;
    const int wm   = (wave & 3) * 32;
    const int wn   = (wave >> 2) * 64;
    const int half = lane >> 4;
    const int l16  = lane & 15;

    v8f acc[2][4];
#pragma unroll
    for (int i = 0; i < 2; ++i)
#pragma unroll
        for (int j = 0; j < 4; ++j) acc[i][j] = (v8f){0.f,0.f,0.f,0.f,0.f,0.f,0.f,0.f};

    const int lastSlot = sBase + cnt - 1;
    const int aRow = tid >> 3;
    const int aK4  = (tid & 7) * 4;
    int myTok[4];
#pragma unroll
    for (int it = 0; it < 4; ++it) {
        int slot = sBase + mBase + aRow + it * 32;
        slot = (slot > lastSlot) ? lastSlot : slot;
        myTok[it] = slot_token[slot];
    }

    for (int kc = 0; kc < DMODEL / 32; ++kc) {
        const int k0 = kc * 32;
#pragma unroll
        for (int it = 0; it < 4; ++it) {
            const float4 v = *(const float4*)(h + (size_t)myTok[it] * DMODEL + k0 + aK4);
            unsigned int* dst = (unsigned int*)(ldsA + (size_t)(aRow + it * 32) * LDT + aK4);
            dst[0] = (unsigned int)f2bf(v.x) | ((unsigned int)f2bf(v.y) << 16);
            dst[1] = (unsigned int)f2bf(v.z) | ((unsigned int)f2bf(v.w) << 16);
        }
#pragma unroll
        for (int it = 0; it < 4; ++it) {
            const int idx = tid + it * 256;
            const int bk  = idx >> 5;
            const int bn  = (idx & 31) * 4;
            const float4 v = *(const float4*)(w1e + (size_t)(k0 + bk) * DFF + nBase + bn);
            ldsB[(size_t)(bn + 0) * LDT + bk] = f2bf(v.x);
            ldsB[(size_t)(bn + 1) * LDT + bk] = f2bf(v.y);
            ldsB[(size_t)(bn + 2) * LDT + bk] = f2bf(v.z);
            ldsB[(size_t)(bn + 3) * LDT + bk] = f2bf(v.w);
        }
        __syncthreads();

        v16bf afr[2], bfr[4];
#pragma unroll
        for (int mi = 0; mi < 2; ++mi) afr[mi] = load_frag_A(ldsA, wm + mi * 16, l16, half);
#pragma unroll
        for (int ni = 0; ni < 4; ++ni) bfr[ni] = load_frag_B(ldsB, wn + ni * 16, l16, half);
#pragma unroll
        for (int mi = 0; mi < 2; ++mi)
#pragma unroll
            for (int ni = 0; ni < 4; ++ni)
                acc[mi][ni] = __builtin_amdgcn_wmma_f32_16x16x32_bf16(
                    false, afr[mi], false, bfr[ni], (short)0, acc[mi][ni], false, false);
        __syncthreads();
    }

#pragma unroll
    for (int mi = 0; mi < 2; ++mi) {
#pragma unroll
        for (int ni = 0; ni < 4; ++ni) {
            const int nG = nBase + wn + ni * 16 + l16;
            const float bias = b1[e * DFF + nG];
#pragma unroll
            for (int r = 0; r < 8; ++r) {
                const int slot = sBase + mBase + wm + mi * 16 + half * 8 + r;
                if (slot <= lastSlot)
                    hid[(size_t)slot * DFF + nG] = f2bf(gelu_tanh(acc[mi][ni][r] + bias));
            }
        }
    }
}

__global__ __launch_bounds__(256)
void moe_gemm2(const unsigned short* __restrict__ hid, const float* __restrict__ w2,
               const float* __restrict__ b2, const int* __restrict__ counts,
               const int* __restrict__ bases, float* __restrict__ ybuf) {
    const int e     = blockIdx.z;
    const int cnt   = counts[e];
    const int mBase = blockIdx.y * 128;
    if (mBase >= cnt) return;
    const int sBase = bases[e];
    const int nBase = blockIdx.x * 128;
    const float* w2e = w2 + (size_t)e * DFF * DMODEL;

    __shared__ unsigned short ldsA[128 * LDT];
    __shared__ unsigned short ldsB[128 * LDT];

    const int tid  = threadIdx.x;
    const int lane = tid & 31;
    const int wave = tid >> 5;
    const int wm   = (wave & 3) * 32;
    const int wn   = (wave >> 2) * 64;
    const int half = lane >> 4;
    const int l16  = lane & 15;

    v8f acc[2][4];
#pragma unroll
    for (int i = 0; i < 2; ++i)
#pragma unroll
        for (int j = 0; j < 4; ++j) acc[i][j] = (v8f){0.f,0.f,0.f,0.f,0.f,0.f,0.f,0.f};

    const int lastSlot = sBase + cnt - 1;
    const int aRow = tid >> 3;
    const int aK4  = (tid & 7) * 4;
    int mySlot[4];
#pragma unroll
    for (int it = 0; it < 4; ++it) {
        int slot = sBase + mBase + aRow + it * 32;
        mySlot[it] = (slot > lastSlot) ? lastSlot : slot;
    }

    for (int kc = 0; kc < DFF / 32; ++kc) {
        const int k0 = kc * 32;
#pragma unroll
        for (int it = 0; it < 4; ++it) {
            const uint2 v = *(const uint2*)(hid + (size_t)mySlot[it] * DFF + k0 + aK4);
            unsigned int* dst = (unsigned int*)(ldsA + (size_t)(aRow + it * 32) * LDT + aK4);
            dst[0] = v.x;
            dst[1] = v.y;
        }
#pragma unroll
        for (int it = 0; it < 4; ++it) {
            const int idx = tid + it * 256;
            const int bk  = idx >> 5;
            const int bn  = (idx & 31) * 4;
            const float4 v = *(const float4*)(w2e + (size_t)(k0 + bk) * DMODEL + nBase + bn);
            ldsB[(size_t)(bn + 0) * LDT + bk] = f2bf(v.x);
            ldsB[(size_t)(bn + 1) * LDT + bk] = f2bf(v.y);
            ldsB[(size_t)(bn + 2) * LDT + bk] = f2bf(v.z);
            ldsB[(size_t)(bn + 3) * LDT + bk] = f2bf(v.w);
        }
        __syncthreads();

        v16bf afr[2], bfr[4];
#pragma unroll
        for (int mi = 0; mi < 2; ++mi) afr[mi] = load_frag_A(ldsA, wm + mi * 16, l16, half);
#pragma unroll
        for (int ni = 0; ni < 4; ++ni) bfr[ni] = load_frag_B(ldsB, wn + ni * 16, l16, half);
#pragma unroll
        for (int mi = 0; mi < 2; ++mi)
#pragma unroll
            for (int ni = 0; ni < 4; ++ni)
                acc[mi][ni] = __builtin_amdgcn_wmma_f32_16x16x32_bf16(
                    false, afr[mi], false, bfr[ni], (short)0, acc[mi][ni], false, false);
        __syncthreads();
    }

#pragma unroll
    for (int mi = 0; mi < 2; ++mi) {
#pragma unroll
        for (int ni = 0; ni < 4; ++ni) {
            const int nG = nBase + wn + ni * 16 + l16;
            const float bias = b2[e * DMODEL + nG];
#pragma unroll
            for (int r = 0; r < 8; ++r) {
                const int slot = sBase + mBase + wm + mi * 16 + half * 8 + r;
                if (slot <= lastSlot)
                    ybuf[(size_t)slot * DMODEL + nG] = acc[mi][ni][r] + bias;
            }
        }
    }
}

// ----------------------------------------------------------------- combine ---
__global__ void moe_combine(const float* __restrict__ ybuf, const int* __restrict__ slot_of,
                            const float* __restrict__ tok_gate, float* __restrict__ out) {
    const int t = blockIdx.x;
    const int s0 = slot_of[t * 2 + 0];
    const int s1 = slot_of[t * 2 + 1];
    const float g0 = tok_gate[t * 2 + 0];
    const float g1 = tok_gate[t * 2 + 1];
    for (int d = threadIdx.x; d < DMODEL; d += blockDim.x)
        out[(size_t)t * DMODEL + d] =
            g0 * ybuf[(size_t)s0 * DMODEL + d] + g1 * ybuf[(size_t)s1 * DMODEL + d];
}

// ------------------------------------------------------------------ launch ---
extern "C" void kernel_launch(void* const* d_in, const int* in_sizes, int n_in,
                              void* d_out, int out_size, void* d_ws, size_t ws_size,
                              hipStream_t stream) {
    const float* hs = (const float*)d_in[0];   // [BT, D]
    const float* rw = (const float*)d_in[1];   // [D, E]
    const float* w1 = (const float*)d_in[2];   // [E, D, DFF]
    const float* b1 = (const float*)d_in[3];   // [E, DFF]
    const float* w2 = (const float*)d_in[4];   // [E, DFF, D]
    const float* b2 = (const float*)d_in[5];   // [E, D]
    float* out = (float*)d_out;

    uintptr_t base = (uintptr_t)d_ws;
    size_t o = 0;
    auto take = [&](size_t bytes) -> void* {
        void* p = (void*)(base + o);
        o = (o + bytes + 255) & ~(size_t)255;
        return p;
    };
    int*   counts     = (int*)take(NEXP * sizeof(int));
    int*   bases      = (int*)take(NEXP * sizeof(int));
    int*   cursors    = (int*)take(NEXP * sizeof(int));
    int*   tok_idx    = (int*)take((size_t)NSLOT * sizeof(int));
    float* tok_gate   = (float*)take((size_t)NSLOT * sizeof(float));
    int*   slot_token = (int*)take((size_t)NSLOT * sizeof(int));
    int*   slot_of    = (int*)take((size_t)NSLOT * sizeof(int));
    unsigned short* hid = (unsigned short*)take((size_t)NSLOT * DFF * sizeof(unsigned short));
    float* ybuf = (float*)take((size_t)NSLOT * DMODEL * sizeof(float));
    // fast-path extras
    unsigned short* hbf = (unsigned short*)take((size_t)BT * DMODEL * sizeof(unsigned short));
    unsigned short* w1t = (unsigned short*)take((size_t)NEXP * DMODEL * DFF * sizeof(unsigned short));
    unsigned short* w2t = (unsigned short*)take((size_t)NEXP * DFF * DMODEL * sizeof(unsigned short));
    const size_t need_big = o;
    (void)in_sizes; (void)n_in; (void)out_size;

    const bool big = (ws_size >= need_big);

    moe_zero<<<1, 32, 0, stream>>>(counts);
    moe_router<<<BT / 8, 256, 0, stream>>>(hs, rw, counts, tok_idx, tok_gate);
    moe_offsets<<<1, 1, 0, stream>>>(counts, bases, cursors);
    moe_scatter<<<NSLOT / 256, 256, 0, stream>>>(tok_idx, cursors, slot_token, slot_of);

    if (big) {
        conv_bf16<<<(BT * DMODEL / 4 + 255) / 256, 256, 0, stream>>>(hs, hbf, BT * DMODEL);
        transp_bf16<<<dim3(DFF / 64, DMODEL / 64, NEXP), 256, 0, stream>>>(w1, w1t, DMODEL, DFF);
        transp_bf16<<<dim3(DMODEL / 64, DFF / 64, NEXP), 256, 0, stream>>>(w2, w2t, DFF, DMODEL);

        dim3 g1(DFF / 128, BT / 128, NEXP);
        moe_gemm1_bf<<<g1, 256, 0, stream>>>(hbf, w1t, b1, counts, bases, slot_token, hid);
        dim3 g2(DMODEL / 128, BT / 128, NEXP);
        moe_gemm2_bf<<<g2, 256, 0, stream>>>(hid, w2t, b2, counts, bases, ybuf);
    } else {
        dim3 g1(DFF / 128, BT / 128, NEXP);
        moe_gemm1<<<g1, 256, 0, stream>>>(hs, w1, b1, counts, bases, slot_token, hid);
        dim3 g2(DMODEL / 128, BT / 128, NEXP);
        moe_gemm2<<<g2, 256, 0, stream>>>(hid, w2, b2, counts, bases, ybuf);
    }

    moe_combine<<<BT, 256, 0, stream>>>(ybuf, slot_of, tok_gate, out);
}
